// SpecialConv_7017976562389
// MI455X (gfx1250) — matrix operand order repaired and checked by reading, not compiled, not run on hardware
//
#include <hip/hip_runtime.h>

typedef float    v4f  __attribute__((ext_vector_type(4)));
typedef float    v8f  __attribute__((ext_vector_type(8)));
typedef _Float16 v8h  __attribute__((ext_vector_type(8)));
typedef _Float16 v16h __attribute__((ext_vector_type(16)));

// Problem sizes
#define BBATCH 1024
#define PP     24
#define HH     2048
#define K1DIM  4096
#define N1DIM  2048
#define K2DIM  2048
#define N2DIM  1024
#define OUTPER 4

// Tiling: block tile 128x64, 8 wave32s in 4x2, each wave 32x32 (4 WMMA accs)
#define BM 128
#define BN 64
#define BK 32
#define LDA 40   // BK + 8 halves pad: 80B row stride, conflict-free, 16B aligned
#define LDB 40
#define NTHREADS 256

__device__ __forceinline__ float gelu_exact(float v) {
    return 0.5f * v * (1.0f + erff(v * 0.70710678118654752440f));
}

__device__ __forceinline__ v16h cat8(v8h lo, v8h hi) {
    return __builtin_shufflevector(lo, hi, 0,1,2,3,4,5,6,7,8,9,10,11,12,13,14,15);
}

__device__ __forceinline__ v8h pack8(v4f a, v4f b) {
    return (v8h){(_Float16)a.x,(_Float16)a.y,(_Float16)a.z,(_Float16)a.w,
                 (_Float16)b.x,(_Float16)b.y,(_Float16)b.z,(_Float16)b.w};
}

__device__ __forceinline__ v8f wmma16(v16h a, v16h b, v8f c) {
    return __builtin_amdgcn_wmma_f32_16x16x32_f16(false, a, false, b, (short)0, c,
                                                  false, false);
}

// ---------------------------------------------------------------------------
// Stage 1: h1[b,p,:] = GELU( x[b, p*H .. p*H+4096) @ W1[p] + b1[p] )
// (window concat of adjacent positions == contiguous slice; p==P-1 tail = 0)
// ---------------------------------------------------------------------------
__global__ __launch_bounds__(NTHREADS) void sc_gemm1_f16(
    const float* __restrict__ x, const float* __restrict__ W1,
    const float* __restrict__ b1, _Float16* __restrict__ h1)
{
    __shared__ _Float16 sA[2][BM * LDA];
    __shared__ _Float16 sB[2][BN * LDB];

    const int p  = blockIdx.z;
    const int n0 = blockIdx.x * BN;
    const int m0 = blockIdx.y * BM;

    const int t    = threadIdx.x;
    const int lane = t & 31;
    const int w    = t >> 5;
    const int wm   = w >> 1;      // 0..3 : 32-row strip
    const int wn   = w & 1;       // 0..1 : 32-col strip
    const int g    = lane >> 4;
    const int l16  = lane & 15;

    // A staging: thread -> (row ra, 16-wide k chunk)
    const int ra = t >> 1;
    const int ka = (t & 1) << 4;
    const float* arow = x + (long)(m0 + ra) * (PP * HH) + (long)p * HH;
    const bool tailzero = (p + 1) >= PP;     // scalar; 2nd K-half must be zero

    // B staging: thread -> (k row kb, 8 consecutive n)
    const int kb    = t >> 3;
    const int nbase = (t & 7) << 3;
    const float* wcol = W1 + (long)p * K1DIM * N1DIM + (long)kb * N1DIM + n0 + nbase;

    v8f acc00 = {}, acc01 = {}, acc10 = {}, acc11 = {};
    v4f A0, A1, A2, A3, Bv0, Bv1;
    const v4f vz = {0.f, 0.f, 0.f, 0.f};

#define G_FETCH(K0)                                                          \
    {                                                                        \
        if (tailzero && (K0) >= HH) {                                        \
            A0 = vz; A1 = vz; A2 = vz; A3 = vz;                              \
        } else {                                                             \
            const v4f* s4_ = (const v4f*)(arow + (K0) + ka);                 \
            A0 = s4_[0]; A1 = s4_[1]; A2 = s4_[2]; A3 = s4_[3];              \
        }                                                                    \
        const v4f* w4_ = (const v4f*)(wcol + (long)(K0) * N1DIM);            \
        Bv0 = w4_[0]; Bv1 = w4_[1];                                          \
    }

#define G_STAGE(BUF)                                                         \
    {                                                                        \
        *(v8h*)&sA[BUF][ra * LDA + ka]     = pack8(A0, A1);                  \
        *(v8h*)&sA[BUF][ra * LDA + ka + 8] = pack8(A2, A3);                  \
        _Float16* bt_ = &sB[BUF][nbase * LDB + kb];                          \
        bt_[0*LDB] = (_Float16)Bv0.x; bt_[1*LDB] = (_Float16)Bv0.y;          \
        bt_[2*LDB] = (_Float16)Bv0.z; bt_[3*LDB] = (_Float16)Bv0.w;          \
        bt_[4*LDB] = (_Float16)Bv1.x; bt_[5*LDB] = (_Float16)Bv1.y;          \
        bt_[6*LDB] = (_Float16)Bv1.z; bt_[7*LDB] = (_Float16)Bv1.w;          \
    }

#define G_COMPUTE(BUF)                                                       \
    {                                                                        \
        const _Float16* bp_ = &sB[BUF][(wn * 32 + l16) * LDB + g * 8];       \
        const v16h bf0_ = cat8(*(const v8h*)bp_, *(const v8h*)(bp_ + 16));   \
        const v16h bf1_ = cat8(*(const v8h*)(bp_ + 16 * LDB),                \
                               *(const v8h*)(bp_ + 16 * LDB + 16));          \
        const _Float16* ap_ = &sA[BUF][(wm * 32 + l16) * LDA + g * 8];       \
        v16h af_ = cat8(*(const v8h*)ap_, *(const v8h*)(ap_ + 16));          \
        acc00 = wmma16(af_, bf0_, acc00);                                    \
        acc01 = wmma16(af_, bf1_, acc01);                                    \
        af_ = cat8(*(const v8h*)(ap_ + 16 * LDA),                            \
                   *(const v8h*)(ap_ + 16 * LDA + 16));                      \
        acc10 = wmma16(af_, bf0_, acc10);                                    \
        acc11 = wmma16(af_, bf1_, acc11);                                    \
    }

    const int NIT = K1DIM / BK;   // 128 (even)
    G_FETCH(0);
    G_STAGE(0);
    __syncthreads();
    for (int it = 0; it < NIT; it += 2) {
        G_FETCH((it + 1) * BK);       // loads in flight across the WMMAs
        G_COMPUTE(0);
        G_STAGE(1);                   // wait + convert after compute
        __syncthreads();
        const bool more = (it + 2) < NIT;
        if (more) G_FETCH((it + 2) * BK);
        G_COMPUTE(1);
        if (more) { G_STAGE(0); __syncthreads(); }
    }
#undef G_FETCH
#undef G_STAGE
#undef G_COMPUTE

    // ---- epilogue: bias + exact GELU -> f16 intermediate ----
    const int   ncol  = n0 + wn * 32 + l16;
    const float bias0 = b1[p * N1DIM + ncol];
    const float bias1 = b1[p * N1DIM + ncol + 16];
    const long  mrow  = m0 + wm * 32 + g * 8;
    _Float16* obase = h1 + (long)p * N1DIM + ncol;
    #pragma unroll
    for (int r = 0; r < 8; ++r) {
        const long r0 = (mrow + r)      * (long)(PP * N1DIM);
        const long r1 = (mrow + 16 + r) * (long)(PP * N1DIM);
        obase[r0]      = (_Float16)gelu_exact(acc00[r] + bias0);
        obase[r0 + 16] = (_Float16)gelu_exact(acc01[r] + bias1);
        obase[r1]      = (_Float16)gelu_exact(acc10[r] + bias0);
        obase[r1 + 16] = (_Float16)gelu_exact(acc11[r] + bias1);
    }
}

// ---------------------------------------------------------------------------
// Stage 2: h2[b,p,:] = GELU( h1[b,p,:] @ W2[p] + b2[p] )   K=2048, N=1024
// ---------------------------------------------------------------------------
__global__ __launch_bounds__(NTHREADS) void sc_gemm2_f16(
    const _Float16* __restrict__ h1, const float* __restrict__ W2,
    const float* __restrict__ b2, _Float16* __restrict__ h2)
{
    __shared__ _Float16 sA[2][BM * LDA];
    __shared__ _Float16 sB[2][BN * LDB];

    const int p  = blockIdx.z;
    const int n0 = blockIdx.x * BN;
    const int m0 = blockIdx.y * BM;

    const int t    = threadIdx.x;
    const int lane = t & 31;
    const int w    = t >> 5;
    const int wm   = w >> 1;
    const int wn   = w & 1;
    const int g    = lane >> 4;
    const int l16  = lane & 15;

    const int ra = t >> 1;
    const int ka = (t & 1) << 4;
    const _Float16* arow = h1 + (long)(m0 + ra) * (PP * K2DIM) + (long)p * K2DIM;

    const int kb    = t >> 3;
    const int nbase = (t & 7) << 3;
    const float* wcol = W2 + (long)p * K2DIM * N2DIM + (long)kb * N2DIM + n0 + nbase;

    v8f acc00 = {}, acc01 = {}, acc10 = {}, acc11 = {};
    v8h aR0, aR1;
    v4f Bv0, Bv1;

#define G_FETCH(K0)                                                          \
    {                                                                        \
        const v8h* a8_ = (const v8h*)(arow + (K0) + ka);                     \
        aR0 = a8_[0]; aR1 = a8_[1];                                          \
        const v4f* w4_ = (const v4f*)(wcol + (long)(K0) * N2DIM);            \
        Bv0 = w4_[0]; Bv1 = w4_[1];                                          \
    }

#define G_STAGE(BUF)                                                         \
    {                                                                        \
        *(v8h*)&sA[BUF][ra * LDA + ka]     = aR0;                            \
        *(v8h*)&sA[BUF][ra * LDA + ka + 8] = aR1;                            \
        _Float16* bt_ = &sB[BUF][nbase * LDB + kb];                          \
        bt_[0*LDB] = (_Float16)Bv0.x; bt_[1*LDB] = (_Float16)Bv0.y;          \
        bt_[2*LDB] = (_Float16)Bv0.z; bt_[3*LDB] = (_Float16)Bv0.w;          \
        bt_[4*LDB] = (_Float16)Bv1.x; bt_[5*LDB] = (_Float16)Bv1.y;          \
        bt_[6*LDB] = (_Float16)Bv1.z; bt_[7*LDB] = (_Float16)Bv1.w;          \
    }

#define G_COMPUTE(BUF)                                                       \
    {                                                                        \
        const _Float16* bp_ = &sB[BUF][(wn * 32 + l16) * LDB + g * 8];       \
        const v16h bf0_ = cat8(*(const v8h*)bp_, *(const v8h*)(bp_ + 16));   \
        const v16h bf1_ = cat8(*(const v8h*)(bp_ + 16 * LDB),                \
                               *(const v8h*)(bp_ + 16 * LDB + 16));          \
        const _Float16* ap_ = &sA[BUF][(wm * 32 + l16) * LDA + g * 8];       \
        v16h af_ = cat8(*(const v8h*)ap_, *(const v8h*)(ap_ + 16));          \
        acc00 = wmma16(af_, bf0_, acc00);                                    \
        acc01 = wmma16(af_, bf1_, acc01);                                    \
        af_ = cat8(*(const v8h*)(ap_ + 16 * LDA),                            \
                   *(const v8h*)(ap_ + 16 * LDA + 16));                      \
        acc10 = wmma16(af_, bf0_, acc10);                                    \
        acc11 = wmma16(af_, bf1_, acc11);                                    \
    }

    const int NIT = K2DIM / BK;   // 64 (even)
    G_FETCH(0);
    G_STAGE(0);
    __syncthreads();
    for (int it = 0; it < NIT; it += 2) {
        G_FETCH((it + 1) * BK);
        G_COMPUTE(0);
        G_STAGE(1);
        __syncthreads();
        const bool more = (it + 2) < NIT;
        if (more) G_FETCH((it + 2) * BK);
        G_COMPUTE(1);
        if (more) { G_STAGE(0); __syncthreads(); }
    }
#undef G_FETCH
#undef G_STAGE
#undef G_COMPUTE

    const int   ncol  = n0 + wn * 32 + l16;
    const float bias0 = b2[p * N2DIM + ncol];
    const float bias1 = b2[p * N2DIM + ncol + 16];
    const long  mrow  = m0 + wm * 32 + g * 8;
    _Float16* obase = h2 + (long)p * N2DIM + ncol;
    #pragma unroll
    for (int r = 0; r < 8; ++r) {
        const long r0 = (mrow + r)      * (long)(PP * N2DIM);
        const long r1 = (mrow + 16 + r) * (long)(PP * N2DIM);
        obase[r0]      = (_Float16)gelu_exact(acc00[r] + bias0);
        obase[r0 + 16] = (_Float16)gelu_exact(acc01[r] + bias1);
        obase[r1]      = (_Float16)gelu_exact(acc10[r] + bias0);
        obase[r1 + 16] = (_Float16)gelu_exact(acc11[r] + bias1);
    }
}

// ---------------------------------------------------------------------------
// Stage 3: out[b, p*4+o] = ReLU( h2[b,p,:] @ W3[p,:,o] + b3[p,o] )
// One wave32 per (b,p); 4 outputs; shuffle reduction.
// ---------------------------------------------------------------------------
__global__ __launch_bounds__(NTHREADS) void sc_stage3(
    const _Float16* __restrict__ h2, const float* __restrict__ W3,
    const float* __restrict__ b3, float* __restrict__ out)
{
    const int gwave = (int)((blockIdx.x * blockDim.x + threadIdx.x) >> 5);
    const int lane  = threadIdx.x & 31;
    if (gwave >= BBATCH * PP) return;
    const int b = gwave / PP;
    const int p = gwave % PP;

    const _Float16* hrow = h2 + ((long)b * PP + p) * (long)N2DIM;
    const float*    wmat = W3 + (long)p * N2DIM * OUTPER;

    float a0 = 0.f, a1 = 0.f, a2 = 0.f, a3 = 0.f;
    for (int i = lane; i < N2DIM; i += 32) {
        const float hv = (float)hrow[i];
        const v4f   wv = *(const v4f*)(wmat + i * OUTPER);
        a0 += hv * wv.x; a1 += hv * wv.y; a2 += hv * wv.z; a3 += hv * wv.w;
    }
    #pragma unroll
    for (int off = 16; off > 0; off >>= 1) {
        a0 += __shfl_xor(a0, off, 32);
        a1 += __shfl_xor(a1, off, 32);
        a2 += __shfl_xor(a2, off, 32);
        a3 += __shfl_xor(a3, off, 32);
    }
    if (lane == 0) {
        const v4f bb = *(const v4f*)(b3 + p * OUTPER);
        float* dst = out + (long)b * (PP * OUTPER) + p * OUTPER;
        dst[0] = fmaxf(a0 + bb.x, 0.f);
        dst[1] = fmaxf(a1 + bb.y, 0.f);
        dst[2] = fmaxf(a2 + bb.z, 0.f);
        dst[3] = fmaxf(a3 + bb.w, 0.f);
    }
}

// ---------------------------------------------------------------------------
extern "C" void kernel_launch(void* const* d_in, const int* in_sizes, int n_in,
                              void* d_out, int out_size, void* d_ws, size_t ws_size,
                              hipStream_t stream) {
    const float* x  = (const float*)d_in[0];
    const float* W1 = (const float*)d_in[1];
    const float* b1 = (const float*)d_in[2];
    const float* W2 = (const float*)d_in[3];
    const float* b2 = (const float*)d_in[4];
    const float* W3 = (const float*)d_in[5];
    const float* b3 = (const float*)d_in[6];
    float* out = (float*)d_out;

    _Float16* h1 = (_Float16*)d_ws;                             // [B, P, 2048] f16
    _Float16* h2 = h1 + (size_t)BBATCH * PP * N1DIM;            // [B, P, 1024] f16

    dim3 g1(N1DIM / BN, BBATCH / BM, PP);   // (32, 8, 24), p slowest => W1 L2 reuse
    sc_gemm1_f16<<<g1, NTHREADS, 0, stream>>>(x, W1, b1, h1);

    dim3 g2(N2DIM / BN, BBATCH / BM, PP);   // (16, 8, 24)
    sc_gemm2_f16<<<g2, NTHREADS, 0, stream>>>(h1, W2, b2, h2);

    const int waves  = BBATCH * PP;                 // 24576
    const int blocks = (waves * 32 + NTHREADS - 1) / NTHREADS;
    sc_stage3<<<blocks, NTHREADS, 0, stream>>>(h2, W3, b3, out);
}
